// BigbirdBlockSpareAttention_87780541596009
// MI455X (gfx1250) — matrix operand order repaired and checked by reading, not compiled, not run on hardware
//
#include <hip/hip_runtime.h>

#define BB 2
#define HH 12
#define SS 4096
#define DH 64
#define BS 64
#define NB 64
#define RR 3
#define NEGV -10000.0f

// padded LDS row strides (72 u16 = 144B = 9*16B: keeps uint4 alignment,
// and lane stride 36 dwords mod 64 banks -> conflict-free fragment loads)
#define DHP 72
#define BSP 72

typedef __attribute__((ext_vector_type(16))) __bf16 v16bf;
typedef __attribute__((ext_vector_type(16))) unsigned short v16u;
typedef __attribute__((ext_vector_type(8))) float v8f;

union Frag {
    v16bf b;
    v16u  u;
    uint4 q[2];
};

__device__ __forceinline__ unsigned short f2bf(float f) {
    unsigned int u = __float_as_uint(f);
    u += 0x7FFFu + ((u >> 16) & 1u);   // round-to-nearest-even bf16
    return (unsigned short)(u >> 16);
}

__global__ __launch_bounds__(128) void bigbird_wmma_kernel(
    const float* __restrict__ Q, const float* __restrict__ K,
    const float* __restrict__ V, const float* __restrict__ Msk,
    const int* __restrict__ Rnd, float* __restrict__ Out) {

    __shared__ __align__(16) unsigned short sK[BS * DHP];     // [key][dim] bf16, padded
    __shared__ __align__(16) unsigned short sVT[DH * BSP];    // [dim][key] bf16, padded
    __shared__ __align__(16) unsigned short sP[4][16 * BSP];  // per wave [qrow][key], padded

    const int wg  = blockIdx.x;
    const int l   = wg % NB;
    const int h   = (wg / NB) % HH;
    const int b   = wg / (NB * HH);
    const int tid  = threadIdx.x;
    const int w    = tid >> 5;        // wave id 0..3
    const int lane = tid & 31;
    const int lm   = lane & 15;
    const int kh   = lane >> 4;       // lane half

    const size_t bh = (size_t)b * HH + h;
    const float* Qbh = Q + bh * SS * DH;
    const float* Kbh = K + bh * SS * DH;
    const float* Vbh = V + bh * SS * DH;
    const float* mb  = Msk + (size_t)b * SS;
    float* Obh       = Out + bh * SS * DH;

    // ---- key-block list (duplicates preserved, as in reference softmax) ----
    int blist[8];
    int buseq[8];
    int nlist;
    bool full = (l == 0) || (l == NB - 1);
    if (full) {
        nlist = NB;
    } else {
        const int* rp = Rnd + (bh * (NB - 2) + (size_t)(l - 1)) * RR;
        if (l == 1) {
            blist[0] = 0; blist[1] = 1; blist[2] = 2; blist[3] = NB - 1;
            buseq[0] = buseq[1] = buseq[2] = buseq[3] = 0;
            blist[4] = rp[0]; blist[5] = rp[1]; blist[6] = rp[2];
            buseq[4] = buseq[5] = buseq[6] = 1;
            nlist = 7;
        } else if (l == NB - 2) {
            blist[0] = 0; blist[1] = NB - 3; blist[2] = NB - 2; blist[3] = NB - 1;
            buseq[0] = buseq[1] = buseq[2] = buseq[3] = 0;
            blist[4] = rp[0]; blist[5] = rp[1]; blist[6] = rp[2];
            buseq[4] = buseq[5] = buseq[6] = 1;
            nlist = 7;
        } else {
            blist[0] = 0;      buseq[0] = 0;
            blist[1] = l - 1;  blist[2] = l; blist[3] = l + 1;
            buseq[1] = buseq[2] = buseq[3] = 1;
            blist[4] = rp[0];  blist[5] = rp[1]; blist[6] = rp[2];
            buseq[4] = buseq[5] = buseq[6] = 1;
            blist[7] = NB - 1; buseq[7] = 0;
            nlist = 8;
        }
    }

    // ---- Q A-fragments (16-bit A 16x32: e<8 -> K=kh*8+e ; e>=8 -> K=16+kh*8+(e-8)) ----
    const int qrow0 = l * BS + w * 16;
    Frag qa[2];
#pragma unroll
    for (int c = 0; c < 2; ++c) {
        const float* qp = Qbh + (size_t)(qrow0 + lm) * DH + c * 32 + kh * 8;
#pragma unroll
        for (int g = 0; g < 2; ++g)
#pragma unroll
            for (int j = 0; j < 8; ++j)
                qa[c].u[g * 8 + j] = f2bf(qp[g * 16 + j]);
    }

    // query-row masks (C-layout row = v + 8*kh)
    float mq[8];
#pragma unroll
    for (int v = 0; v < 8; ++v) mq[v] = mb[qrow0 + v + 8 * kh];

    // ---- flash accumulators ----
    v8f Oacc[4];
#pragma unroll
    for (int t = 0; t < 4; ++t)
#pragma unroll
        for (int v = 0; v < 8; ++v) Oacc[t][v] = 0.0f;
    float mrow[8], ssum[8];
#pragma unroll
    for (int v = 0; v < 8; ++v) { mrow[v] = -3.0e38f; ssum[v] = 0.0f; }

    const float scale = 0.125f;  // 1/sqrt(64)
    unsigned short* pw = sP[w];

    for (int it = 0; it < nlist; ++it) {
        const int kb   = full ? it : blist[it];
        const int useq = full ? 0 : buseq[it];

        // ---- stage K and V^T into LDS as bf16 ----
        // one key row per thread-half: rows consecutive within a wave
        {
            const int row = tid & 63;
            const int d0  = (tid >> 6) * 32;
            const float* kp = Kbh + ((size_t)kb * BS + row) * DH + d0;
            const float* vp = Vbh + ((size_t)kb * BS + row) * DH + d0;
            unsigned int* sK32 = (unsigned int*)&sK[row * DHP + d0];
#pragma unroll
            for (int i = 0; i < 16; ++i) {
                const unsigned int lo = f2bf(kp[2 * i]);
                const unsigned int hi = f2bf(kp[2 * i + 1]);
                sK32[i] = lo | (hi << 16);
            }
#pragma unroll
            for (int i = 0; i < 32; ++i)
                sVT[(d0 + i) * BSP + row] = f2bf(vp[i]);
        }
        __syncthreads();

        // prefetch next key block (global_prefetch_b8)
        if (it + 1 < nlist) {
            const int nkb = full ? (it + 1) : blist[it + 1];
            const int row = tid & 63;
            __builtin_prefetch(Kbh + ((size_t)nkb * BS + row) * DH, 0, 1);
            __builtin_prefetch(Vbh + ((size_t)nkb * BS + row) * DH, 0, 1);
        }

        // key masks per column tile
        float mk[4];
#pragma unroll
        for (int t = 0; t < 4; ++t) mk[t] = mb[kb * BS + t * 16 + lm];

        // ---- preload all K B-fragments, then back-to-back WMMAs ----
        Frag kf[4][2];
#pragma unroll
        for (int t = 0; t < 4; ++t)
#pragma unroll
            for (int c = 0; c < 2; ++c) {
                const int base = (t * 16 + lm) * DHP + c * 32 + kh * 8;
                kf[t][c].q[0] = *(const uint4*)&sK[base];
                kf[t][c].q[1] = *(const uint4*)&sK[base + 16];
            }

        v8f st[4];
#pragma unroll
        for (int t = 0; t < 4; ++t) {
            v8f acc = {0.f, 0.f, 0.f, 0.f, 0.f, 0.f, 0.f, 0.f};
            acc = __builtin_amdgcn_wmma_f32_16x16x32_bf16(
                false, qa[0].b, false, kf[t][0].b, (short)0, acc, false, false);
            acc = __builtin_amdgcn_wmma_f32_16x16x32_bf16(
                false, qa[1].b, false, kf[t][1].b, (short)0, acc, false, false);
            st[t] = acc;
        }

        // ---- mask + scale ----
        float lgt[4][8];
#pragma unroll
        for (int t = 0; t < 4; ++t)
#pragma unroll
            for (int v = 0; v < 8; ++v) {
                const float m = useq ? (mk[t] * mq[v]) : mk[t];
                lgt[t][v] = st[t][v] * scale + (1.0f - m) * NEGV;
            }

        // ---- online softmax (row across 16 lanes + 4 tiles) ----
#pragma unroll
        for (int v = 0; v < 8; ++v) {
            float tv = fmaxf(fmaxf(lgt[0][v], lgt[1][v]), fmaxf(lgt[2][v], lgt[3][v]));
#pragma unroll
            for (int o = 1; o < 16; o <<= 1) tv = fmaxf(tv, __shfl_xor(tv, o, 32));
            const float nm  = fmaxf(mrow[v], tv);
            const float fsc = __expf(mrow[v] - nm);
            ssum[v] *= fsc;
#pragma unroll
            for (int t = 0; t < 4; ++t) Oacc[t][v] *= fsc;
            float rs = 0.0f;
#pragma unroll
            for (int t = 0; t < 4; ++t) {
                const float p = __expf(lgt[t][v] - nm);
                lgt[t][v] = p;
                rs += p;
            }
#pragma unroll
            for (int o = 1; o < 16; o <<= 1) rs += __shfl_xor(rs, o, 32);
            ssum[v] += rs;
            mrow[v] = nm;
        }

        // ---- P: C-layout -> LDS -> A-fragments (bf16) ----
#pragma unroll
        for (int t = 0; t < 4; ++t)
#pragma unroll
            for (int v = 0; v < 8; ++v)
                pw[(v + 8 * kh) * BSP + t * 16 + lm] = f2bf(lgt[t][v]);

        // ---- O += P * V : preload fragments, then back-to-back WMMAs ----
#pragma unroll
        for (int c = 0; c < 2; ++c) {
            Frag pf;  // A layout
            const int pb = lm * BSP + c * 32 + kh * 8;
            pf.q[0] = *(const uint4*)&pw[pb];
            pf.q[1] = *(const uint4*)&pw[pb + 16];
            Frag vf[4];  // B layout from transposed V
#pragma unroll
            for (int td = 0; td < 4; ++td) {
                const int vb = (td * 16 + lm) * BSP + c * 32 + kh * 8;
                vf[td].q[0] = *(const uint4*)&sVT[vb];
                vf[td].q[1] = *(const uint4*)&sVT[vb + 16];
            }
#pragma unroll
            for (int td = 0; td < 4; ++td)
                Oacc[td] = __builtin_amdgcn_wmma_f32_16x16x32_bf16(
                    false, pf.b, false, vf[td].b, (short)0, Oacc[td], false, false);
        }
        __syncthreads();
    }

    // ---- normalize, apply from-mask, write out ----
#pragma unroll
    for (int v = 0; v < 8; ++v) {
        const int qr = qrow0 + v + 8 * kh;
        const float inv = 1.0f / ssum[v];
        const float fm  = mq[v];
#pragma unroll
        for (int t = 0; t < 4; ++t)
            Obh[(size_t)qr * DH + t * 16 + lm] = Oacc[t][v] * inv * fm;
    }
}

extern "C" void kernel_launch(void* const* d_in, const int* in_sizes, int n_in,
                              void* d_out, int out_size, void* d_ws, size_t ws_size,
                              hipStream_t stream) {
    const float* q = (const float*)d_in[0];
    const float* k = (const float*)d_in[1];
    const float* v = (const float*)d_in[2];
    const float* m = (const float*)d_in[3];
    const int*   r = (const int*)d_in[4];
    float* out = (float*)d_out;

    dim3 grid(BB * HH * NB);   // one workgroup per (b, h, query-block)
    dim3 block(128);           // 4 wave32s, one 16-row strip each
    bigbird_wmma_kernel<<<grid, block, 0, stream>>>(q, k, v, m, r, out);
}
